// Embedding_layer_14147622273701
// MI455X (gfx1250) — matrix operand order repaired and checked by reading, not compiled, not run on hardware
//
#include <hip/hip_runtime.h>
#include <hip/hip_bf16.h>

// ---------------- problem constants (from the reference) ----------------
#define SENT     256
#define BATCH    64
#define WLEN     16
#define D_WORD   300
#define D_CHAR   50
#define OUT_CH   200
#define KSIZE    3
#define KDIM     (D_CHAR * KSIZE)       // 150
#define KPAD     160                    // 5 chunks of K=32 for wmma f16
#define NCHUNK   5
#define NWORDS   (SENT * BATCH)         // 16384
#define OUT_STRIDE (D_WORD + OUT_CH)    // 500
#define TVALID   (WLEN - KSIZE + 1)     // 14 conv output positions

#define NWAVES   7                      // each wave owns TWO 16-col oc tiles
#define BLOCK2   (NWAVES * 32)          // 224 threads (14 tiles >= 13 needed)

typedef __attribute__((ext_vector_type(16))) _Float16 v16h;
typedef __attribute__((ext_vector_type(8)))  _Float16 v8h;
typedef __attribute__((ext_vector_type(8)))  float    v8f;

// ---------------------------------------------------------------------------
// Kernel 1: word embedding gather.  out[:, :, 0:300] = W_word[words]
// One wave per (s,b) row; 75 x float4 per row (all 16B aligned).
// ---------------------------------------------------------------------------
__global__ __launch_bounds__(256) void word_emb_gather(
    const int* __restrict__ words,      // [S, B]
    const float* __restrict__ W_word,   // [V_WORD, 300]
    float* __restrict__ out)            // [S, B, 500]
{
    const int row  = blockIdx.x * 8 + (threadIdx.x >> 5);   // s*64 + b
    const int lane = threadIdx.x & 31;
    const int widx = words[row];
    const float4* __restrict__ src = (const float4*)(W_word + (size_t)widx * D_WORD);
    float4* __restrict__ dst       = (float4*)(out + (size_t)row * OUT_STRIDE);
    #pragma unroll
    for (int i = lane; i < D_WORD / 4; i += 32)   // 75 float4
        dst[i] = src[i];
}

// ---------------------------------------------------------------------------
// Kernel 2: char CNN as a WMMA GEMM per word, transposed so max-pool is an
// in-lane reduction, with TWO output tiles per wave so each LDS im2col read
// feeds two WMMAs:
//   D[t, oc] = sum_K  A'[t,K] * B'[K,oc]
//   A' = f16 im2col of this word (LDS, shared);  B' = conv_w^T (registers)
//   out[:, :, 300:500] = max_{t<14} D[t, :] + bias
//
// A operand (16-bit, 16x32): lane M=t=lane&15; halves = two contiguous runs
//   K = 32c+8g+0..7 and K = 32c+16+8g+0..7 -> two 16B ds_load_b128 from the
//   LDS im2col IH[t][K] (row stride 160 halves = 320 B). All 5 chunks are
//   loaded before the WMMA chain so the ds clause overlaps the waits.
// B operand (16-bit, 32x16): lane N=oc; halves K = 32c+8g+0..7 and 32c+16+8g+0..7 (as A) -> two
//   guarded runs of conv_w[oc*150 + K], loaded once per tile.
// D (f32 16x16): VGPR r holds row t = r + 8g, so max over t is 7 in-lane
//   v_max (+masks for t=14,15) + ONE shfl_xor(16) merge per tile.
// ---------------------------------------------------------------------------
__global__ __launch_bounds__(BLOCK2) void char_cnn_wmma(
    const int* __restrict__ chars,      // [B, S, 16]
    const float* __restrict__ W_char,   // [128, 50]
    const float* __restrict__ conv_w,   // [200, 50, 3] -> row-major [200,150]
    const float* __restrict__ conv_b,   // [200]
    float* __restrict__ out)            // [S, B, 500]
{
    __shared__ float Es[WLEN * D_CHAR];                    // [l, c]  (3.2 KB)
    __shared__ __align__(32) _Float16 IH[16 * KPAD];       // [t, K]  (5.0 KB)

    const int tid  = threadIdx.x;
    const int lane = tid & 31;
    const int wave = tid >> 5;          // owns oc tiles 2*wave, 2*wave+1
    const int g    = lane >> 4;         // half-wave group
    const int tcol = lane & 15;

    // ---- preload B operands (conv weights^T) + biases for both tiles ----
    const int oc0 = wave * 32 + tcol;           // tile 2*wave
    const int oc1 = oc0 + 16;                   // tile 2*wave+1
    const bool oc0v = (oc0 < OUT_CH);
    const bool oc1v = (oc1 < OUT_CH);
    const float bias0 = oc0v ? conv_b[oc0] : 0.0f;
    const float bias1 = oc1v ? conv_b[oc1] : 0.0f;
    v16h bw0[NCHUNK], bw1[NCHUNK];
    #pragma unroll
    for (int c = 0; c < NCHUNK; ++c) {
        const int K0 = c * 32 + 8 * g;
        #pragma unroll
        for (int h = 0; h < 16; ++h) {
            const int K = K0 + h + (h >= 8 ? 8 : 0);
            const bool kv = (K < KDIM);
            bw0[c][h] = (_Float16)((oc0v && kv) ? conv_w[oc0 * KDIM + K] : 0.0f);
            bw1[c][h] = (_Float16)((oc1v && kv) ? conv_w[oc1 * KDIM + K] : 0.0f);
        }
    }

    for (int w = blockIdx.x; w < NWORDS; w += gridDim.x) {
        const int s = w >> 6;           // word (s, b); out row index == w
        const int b = w & 63;
        const int charsBase = (b * SENT + s) * WLEN;

        __syncthreads();                // prev-iteration readers done with Es/IH

        // ---- stage 1: gather char embedding rows into LDS ----
        for (int i = tid; i < WLEN * D_CHAR; i += BLOCK2) {
            const int r = i / D_CHAR;
            const int c = i - r * D_CHAR;
            const int ci = chars[charsBase + r];
            Es[i] = W_char[ci * D_CHAR + c];
        }
        __syncthreads();

        // ---- stage 2: f16 im2col IH[t][K] = E[t + K%3, K/3], zero padded ----
        for (int i = tid; i < 16 * KPAD; i += BLOCK2) {
            const int t = i / KPAD;
            const int K = i - t * KPAD;
            float v = 0.0f;
            if (K < KDIM && t < TVALID) {
                const int ch = K / KSIZE;
                const int k  = K - ch * KSIZE;
                v = Es[(t + k) * D_CHAR + ch];
            }
            IH[i] = (_Float16)v;
        }
        __syncthreads();

        // ---- stage 3: load all A chunks, then the 10-WMMA chain ----
        const _Float16* ap = &IH[tcol * KPAD + 8 * g];  // lane row t = tcol
        v16h av[NCHUNK];
        #pragma unroll
        for (int c = 0; c < NCHUNK; ++c) {
            const v8h lo = *(const v8h*)(ap + c * 32);        // K=32c+8g+0..7
            const v8h hi = *(const v8h*)(ap + c * 32 + 16);   // K=32c+16+8g+..
            av[c] = __builtin_shufflevector(lo, hi,
                0, 1, 2, 3, 4, 5, 6, 7, 8, 9, 10, 11, 12, 13, 14, 15);
        }
        v8f acc0 = {0.f, 0.f, 0.f, 0.f, 0.f, 0.f, 0.f, 0.f};
        v8f acc1 = {0.f, 0.f, 0.f, 0.f, 0.f, 0.f, 0.f, 0.f};
        #pragma unroll
        for (int c = 0; c < NCHUNK; ++c) {
            acc0 = __builtin_amdgcn_wmma_f32_16x16x32_f16(
                       false, av[c], false, bw0[c], (short)0, acc0, false, false);
            acc1 = __builtin_amdgcn_wmma_f32_16x16x32_f16(
                       false, av[c], false, bw1[c], (short)0, acc1, false, false);
        }

        // ---- stage 4: in-lane max over t = r + 8g (valid t < 14) ----
        float x0 = acc0[0], x1 = acc1[0];        // t = 8g, always valid
        #pragma unroll
        for (int r = 1; r < 8; ++r) {
            const bool tv = ((r + 8 * g) < TVALID);
            x0 = fmaxf(x0, tv ? acc0[r] : -3.402823466e38f);
            x1 = fmaxf(x1, tv ? acc1[r] : -3.402823466e38f);
        }
        x0 = fmaxf(x0, __shfl_xor(x0, 16, 32));  // merge t 0..7 with 8..15
        x1 = fmaxf(x1, __shfl_xor(x1, 16, 32));

        float* __restrict__ orow = out + (size_t)w * OUT_STRIDE + D_WORD;
        if (g == 0 && oc0v) orow[oc0] = x0 + bias0;   // coalesced 16-lane store
        if (g == 0 && oc1v) orow[oc1] = x1 + bias1;
    }
}

// ---------------------------------------------------------------------------
extern "C" void kernel_launch(void* const* d_in, const int* in_sizes, int n_in,
                              void* d_out, int out_size, void* d_ws, size_t ws_size,
                              hipStream_t stream) {
    const int*   words  = (const int*)  d_in[0];
    const int*   chars  = (const int*)  d_in[1];
    const float* W_word = (const float*)d_in[2];
    const float* W_char = (const float*)d_in[3];
    const float* conv_w = (const float*)d_in[4];
    const float* conv_b = (const float*)d_in[5];
    float* out = (float*)d_out;

    word_emb_gather<<<NWORDS / 8, 256, 0, stream>>>(words, W_word, out);
    char_cnn_wmma<<<1024, BLOCK2, 0, stream>>>(chars, W_char, conv_w, conv_b, out);
}
